// QuantAttention_64269890617868
// MI455X (gfx1250) — 16-bit and 8-bit matrix operand order repaired and checked by reading, compiled, not run on hardware
//
#include <hip/hip_runtime.h>
#include <hip/hip_bf16.h>


typedef __attribute__((ext_vector_type(16))) _Float16 v16h;
typedef __attribute__((ext_vector_type(2)))  _Float16 h2_t;
typedef __attribute__((ext_vector_type(8)))  float    v8f;
typedef __attribute__((ext_vector_type(8)))  int      v8i;

constexpr int kLTXT = 256;
constexpr int kLIMG = 2048;
constexpr int kS    = 2304;
constexpr int kD    = 2048;
constexpr int kH    = 16;
constexpr int kDH   = 128;
constexpr float kQMAX = 127.0f;
constexpr float kSM_SCALE = 0.08838834764831845f;

__device__ __forceinline__ float dq_scale(unsigned bits) {
    return fmaxf(__uint_as_float(bits) / kQMAX, 1e-8f);
}

__device__ __forceinline__ void async_copy_b128(void* lds_ptr, const void* gptr) {
    unsigned loff = (unsigned)(unsigned long long)lds_ptr;
    asm volatile("global_load_async_to_lds_b128 %0, %1, off"
                 :: "v"(loff), "v"(gptr)
                 : "memory");
}
__device__ __forceinline__ void wait_async_le4() {
    asm volatile("s_wait_asynccnt 0x4" ::: "memory");
}
__device__ __forceinline__ void wait_async_0() {
    asm volatile("s_wait_asynccnt 0x0" ::: "memory");
}

__global__ void absmax_kernel(const float* __restrict__ X, int n, int boundary,
                              unsigned* __restrict__ amax_a,
                              unsigned* __restrict__ amax_b) {
    int i = blockIdx.x * 256 + threadIdx.x;
    float a = 0.0f, b = 0.0f;
    if (i < n) {
        float v = fabsf(X[i]);
        if (i < boundary) b = v; else a = v;
    }
    #pragma unroll
    for (int d = 1; d < 32; d <<= 1) {
        a = fmaxf(a, __shfl_xor(a, d));
        b = fmaxf(b, __shfl_xor(b, d));
    }
    if ((threadIdx.x & 31) == 0) {
        atomicMax(amax_a, __float_as_uint(a));
        atomicMax(amax_b, __float_as_uint(b));
    }
}

__global__ void quant_i8_kernel(const float* __restrict__ X,
                                const unsigned* __restrict__ amax,
                                signed char* __restrict__ Q, int n) {
    int i = blockIdx.x * 256 + threadIdx.x;
    if (i >= n) return;
    float inv = 1.0f / dq_scale(*amax);
    float q = rintf(X[i] * inv);
    q = fminf(fmaxf(q, -kQMAX), kQMAX);
    Q[i] = (signed char)q;
}

__global__ __launch_bounds__(256)
void quant_w_kernel(const float* __restrict__ W, signed char* __restrict__ W8,
                    float* __restrict__ wscale, int N, int K) {
    int wave = threadIdx.x >> 5, lane = threadIdx.x & 31;
    int row = blockIdx.x * 8 + wave;
    if (row >= N) return;
    const float* src = W + (size_t)row * K;
    float mx = 0.0f;
    for (int k = lane; k < K; k += 32) mx = fmaxf(mx, fabsf(src[k]));
    #pragma unroll
    for (int d = 1; d < 32; d <<= 1) mx = fmaxf(mx, __shfl_xor(mx, d));
    float scale = fmaxf(mx / kQMAX, 1e-8f);
    float inv = 1.0f / scale;
    signed char* dst = W8 + (size_t)row * K;
    for (int k = lane; k < K; k += 32) {
        float q = rintf(src[k] * inv);
        q = fminf(fmaxf(q, -kQMAX), kQMAX);
        dst[k] = (signed char)q;
    }
    if (lane == 0) wscale[row] = scale;
}

constexpr int kTK   = 64;
constexpr int kLST  = 80;

__global__ __launch_bounds__(256)
void gemm_i8_kernel(const signed char* __restrict__ A8,
                    const signed char* __restrict__ W8,
                    const float* __restrict__ wscale,
                    const float* __restrict__ bias,
                    const unsigned* __restrict__ amax,
                    float* __restrict__ Out,
                    int M, int N, int K) {
    __shared__ signed char ldsA[2][128 * kLST];
    __shared__ signed char ldsB[2][128 * kLST];

    const int tid   = threadIdx.x;
    const int wave  = tid >> 5;
    const int lane  = tid & 31;
    const int lhalf = lane >> 4;
    const int l16   = lane & 15;
    const int row0  = blockIdx.x * 128;
    const int col0  = blockIdx.y * 128;

    auto issue_stage = [&](int buf, int k0) {
        #pragma unroll
        for (int j = 0; j < 2; ++j) {
            int s = tid + 256 * j;
            int r = s >> 2, c = (s & 3) * 16;
            async_copy_b128(&ldsA[buf][r * kLST + c],
                            A8 + (size_t)(row0 + r) * K + k0 + c);
        }
        #pragma unroll
        for (int j = 0; j < 2; ++j) {
            int s = tid + 256 * j;
            int r = s >> 2, c = (s & 3) * 16;
            async_copy_b128(&ldsB[buf][r * kLST + c],
                            W8 + (size_t)(col0 + r) * K + k0 + c);
        }
    };

    v8i acc[8];
    #pragma unroll
    for (int t = 0; t < 8; ++t) acc[t] = (v8i){0,0,0,0,0,0,0,0};

    const int ksteps = K / kTK;
    issue_stage(0, 0);

    for (int i = 0; i < ksteps; ++i) {
        const int cur = i & 1;
        if (i + 1 < ksteps) {
            issue_stage(cur ^ 1, (i + 1) * kTK);
            wait_async_le4();
        } else {
            wait_async_0();
        }
        __syncthreads();

        const signed char* bA = ldsA[cur];
        const signed char* bB = ldsB[cur];

        v8i af;
        #pragma unroll
        for (int v = 0; v < 8; ++v) {
            int kk = (v >> 1) * 16 + (v & 1) * 4 + lhalf * 8;
            af[v] = *(const int*)(bA + (wave * 16 + l16) * kLST + kk);
        }
        v8i bfr[8];
        #pragma unroll
        for (int t = 0; t < 8; ++t) {
            #pragma unroll
            for (int v = 0; v < 8; ++v) {
                int kk = (v >> 1) * 16 + (v & 1) * 4 + lhalf * 8;
                bfr[t][v] = *(const int*)(bB + (t * 16 + l16) * kLST + kk);
            }
        }
        __builtin_amdgcn_sched_barrier(0);
        #pragma unroll
        for (int t = 0; t < 8; ++t)
            acc[t] = __builtin_amdgcn_wmma_i32_16x16x64_iu8(
                true, af, true, bfr[t], acc[t], false, false);
        __builtin_amdgcn_sched_barrier(0);

        __syncthreads();
    }

    const float sa = dq_scale(*amax);
    #pragma unroll
    for (int t = 0; t < 8; ++t) {
        #pragma unroll
        for (int r = 0; r < 8; ++r) {
            int row = row0 + wave * 16 + r + lhalf * 8;
            int col = col0 + t * 16 + l16;
            Out[(size_t)row * N + col] =
                (float)acc[t][r] * (sa * wscale[col]) + bias[col];
        }
    }
}

__global__ __launch_bounds__(256)
void rmsnorm_absmax_kernel(float* __restrict__ X,
                           const float* __restrict__ w_img,
                           const float* __restrict__ w_enc,
                           unsigned* __restrict__ amax_img,
                           unsigned* __restrict__ amax_enc, int nrows) {
    int wave = threadIdx.x >> 5, lane = threadIdx.x & 31;
    int row = blockIdx.x * 8 + wave;
    if (row >= nrows) return;
    int s = row >> 4;
    const float* w = (s < kLTXT) ? w_enc : w_img;
    float* x = X + (size_t)row * kDH;

    float v[4], ss = 0.0f;
    #pragma unroll
    for (int i = 0; i < 4; ++i) { v[i] = x[lane * 4 + i]; ss += v[i] * v[i]; }
    #pragma unroll
    for (int d = 1; d < 32; d <<= 1) ss += __shfl_xor(ss, d);
    float r = rsqrtf(ss * (1.0f / kDH) + 1e-6f);

    float mx = 0.0f;
    #pragma unroll
    for (int i = 0; i < 4; ++i) {
        float y = v[i] * r * w[lane * 4 + i];
        x[lane * 4 + i] = y;
        mx = fmaxf(mx, fabsf(y));
    }
    #pragma unroll
    for (int d = 1; d < 32; d <<= 1) mx = fmaxf(mx, __shfl_xor(mx, d));
    if (lane == 0)
        atomicMax((s < kLTXT) ? amax_enc : amax_img, __float_as_uint(mx));
}

__global__ void quant_rope_kernel(const float* __restrict__ X,
                                  const float* __restrict__ cosb,
                                  const float* __restrict__ sinb,
                                  const unsigned* __restrict__ amax_img,
                                  const unsigned* __restrict__ amax_enc,
                                  _Float16* __restrict__ Outh) {
    int p = blockIdx.x * 256 + threadIdx.x;
    if (p >= kS * (kD / 2)) return;
    int s = p >> 10;
    int f0 = (p & 1023) * 2;
    float scale = dq_scale((s < kLTXT) ? *amax_enc : *amax_img);
    float inv = 1.0f / scale;
    float x0 = X[(size_t)s * kD + f0];
    float x1 = X[(size_t)s * kD + f0 + 1];
    float q0 = fminf(fmaxf(rintf(x0 * inv), -kQMAX), kQMAX) * scale;
    float q1 = fminf(fmaxf(rintf(x1 * inv), -kQMAX), kQMAX) * scale;
    int d0 = f0 & (kDH - 1);
    float c0 = cosb[s * kDH + d0],     sn0 = sinb[s * kDH + d0];
    float c1 = cosb[s * kDH + d0 + 1], sn1 = sinb[s * kDH + d0 + 1];
    float y0 = q0 * c0 - q1 * sn0;
    float y1 = q1 * c1 + q0 * sn1;
    Outh[(size_t)s * kD + f0]     = (_Float16)y0;
    Outh[(size_t)s * kD + f0 + 1] = (_Float16)y1;
}

__global__ void quant_v_kernel(const float* __restrict__ X,
                               const unsigned* __restrict__ amax_img,
                               const unsigned* __restrict__ amax_enc,
                               _Float16* __restrict__ Vt) {
    int i = blockIdx.x * 256 + threadIdx.x;
    if (i >= kS * kD) return;
    int s = i >> 11;
    int f = i & (kD - 1);
    float scale = dq_scale((s < kLTXT) ? *amax_enc : *amax_img);
    float inv = 1.0f / scale;
    float q = fminf(fmaxf(rintf(X[i] * inv), -kQMAX), kQMAX) * scale;
    Vt[(size_t)f * kS + s] = (_Float16)q;
}

__global__ __launch_bounds__(128)
void flash_attn_kernel(const _Float16* __restrict__ Qh,
                       const _Float16* __restrict__ Kh,
                       const _Float16* __restrict__ Vt,
                       float* __restrict__ Out) {
    const int h     = blockIdx.y;
    const int wave  = threadIdx.x >> 5;
    const int lane  = threadIdx.x & 31;
    const int lhalf = lane >> 4;
    const int l16   = lane & 15;
    const int qrow0 = blockIdx.x * 64 + wave * 16;

    __shared__ _Float16 plds[4][16 * 32];
    _Float16* pbuf = plds[wave];

    v16h qf[4];
    const _Float16* qbase = Qh + (size_t)(qrow0 + l16) * kD + h * kDH;
    #pragma unroll
    for (int c = 0; c < 4; ++c) {
        #pragma unroll
        for (int v = 0; v < 8; ++v) {
            int kd = c * 32 + lhalf * 16 + 2 * v;
            h2_t p = *(const h2_t*)(qbase + kd);
            qf[c][2 * v] = p[0]; qf[c][2 * v + 1] = p[1];
        }
    }

    v8f acc[8];
    float m_i[8], l_i[8];
    #pragma unroll
    for (int t = 0; t < 8; ++t) acc[t] = (v8f){0,0,0,0,0,0,0,0};
    #pragma unroll
    for (int r = 0; r < 8; ++r) { m_i[r] = -3.0e38f; l_i[r] = 0.0f; }

    for (int j = 0; j < kS / 32; ++j) {
        v8f st[2];
        #pragma unroll
        for (int tt = 0; tt < 2; ++tt) {
            v8f c = (v8f){0,0,0,0,0,0,0,0};
            const _Float16* kbase =
                Kh + (size_t)(j * 32 + tt * 16 + l16) * kD + h * kDH;
            #pragma unroll
            for (int cc = 0; cc < 4; ++cc) {
                v16h bf;
                #pragma unroll
                for (int v = 0; v < 8; ++v) {
                    h2_t p = *(const h2_t*)(kbase + cc * 32 + lhalf * 16 + 2 * v);
                    bf[2 * v] = p[0]; bf[2 * v + 1] = p[1];
                }
                c = __builtin_amdgcn_wmma_f32_16x16x32_f16(
                        false, qf[cc], false, bf, (short)0, c, false, false);
            }
            st[tt] = c;
        }

        float alpha[8];
        #pragma unroll
        for (int r = 0; r < 8; ++r) {
            float s0 = st[0][r] * kSM_SCALE;
            float s1 = st[1][r] * kSM_SCALE;
            float mx = fmaxf(s0, s1);
            mx = fmaxf(mx, __shfl_xor(mx, 1));
            mx = fmaxf(mx, __shfl_xor(mx, 2));
            mx = fmaxf(mx, __shfl_xor(mx, 4));
            mx = fmaxf(mx, __shfl_xor(mx, 8));
            float mnew = fmaxf(m_i[r], mx);
            float a = __expf(m_i[r] - mnew);
            float p0 = __expf(s0 - mnew);
            float p1 = __expf(s1 - mnew);
            st[0][r] = p0; st[1][r] = p1;
            float rs = p0 + p1;
            rs += __shfl_xor(rs, 1);
            rs += __shfl_xor(rs, 2);
            rs += __shfl_xor(rs, 4);
            rs += __shfl_xor(rs, 8);
            l_i[r] = l_i[r] * a + rs;
            m_i[r] = mnew;
            alpha[r] = a;
        }
        #pragma unroll
        for (int t = 0; t < 8; ++t)
            #pragma unroll
            for (int r = 0; r < 8; ++r)
                acc[t][r] *= alpha[r];

        #pragma unroll
        for (int tt = 0; tt < 2; ++tt)
            #pragma unroll
            for (int r = 0; r < 8; ++r)
                pbuf[(r + lhalf * 8) * 32 + tt * 16 + l16] = (_Float16)st[tt][r];
        v16h pf;
        #pragma unroll
        for (int v = 0; v < 8; ++v) {
            int kd = lhalf * 16 + 2 * v;
            h2_t p = *(const h2_t*)(pbuf + l16 * 32 + kd);
            pf[2 * v] = p[0]; pf[2 * v + 1] = p[1];
        }

        #pragma unroll
        for (int t = 0; t < 8; ++t) {
            const _Float16* vbase =
                Vt + (size_t)(h * kDH + t * 16 + l16) * kS + j * 32 + lhalf * 16;
            v16h vf;
            #pragma unroll
            for (int v = 0; v < 8; ++v) {
                h2_t p = *(const h2_t*)(vbase + 2 * v);
                vf[2 * v] = p[0]; vf[2 * v + 1] = p[1];
            }
            acc[t] = __builtin_amdgcn_wmma_f32_16x16x32_f16(
                        false, pf, false, vf, (short)0, acc[t], false, false);
        }
    }

    #pragma unroll
    for (int t = 0; t < 8; ++t)
        #pragma unroll
        for (int r = 0; r < 8; ++r) {
            int row = qrow0 + r + lhalf * 8;
            int col = h * kDH + t * 16 + l16;
            Out[(size_t)row * kD + col] = acc[t][r] / l_i[r];
        }
}

extern "C" void kernel_launch(void* const* d_in, const int* in_sizes, int n_in,
                              void* d_out, int out_size, void* d_ws, size_t ws_size,
                              hipStream_t stream) {
    const float* hidden   = (const float*)d_in[0];
    const float* encoder  = (const float*)d_in[1];
    const float* rope_cos = (const float*)d_in[2];
    const float* rope_sin = (const float*)d_in[3];
    const float* Wf[8] = { (const float*)d_in[4],  (const float*)d_in[6],
                           (const float*)d_in[8],  (const float*)d_in[10],
                           (const float*)d_in[12], (const float*)d_in[14],
                           (const float*)d_in[16], (const float*)d_in[18] };
    const float* Bf[8] = { (const float*)d_in[5],  (const float*)d_in[7],
                           (const float*)d_in[9],  (const float*)d_in[11],
                           (const float*)d_in[13], (const float*)d_in[15],
                           (const float*)d_in[17], (const float*)d_in[19] };
    const float* norm_q  = (const float*)d_in[20];
    const float* norm_k  = (const float*)d_in[21];
    const float* norm_aq = (const float*)d_in[22];
    const float* norm_ak = (const float*)d_in[23];

    constexpr size_t SZ_W    = (size_t)kD * kD;
    constexpr size_t OFF_WS  = 256;
    constexpr size_t OFF_H8  = OFF_WS + 8 * kD * sizeof(float);
    constexpr size_t OFF_E8  = OFF_H8 + (size_t)kLIMG * kD;
    constexpr size_t OFF_W8  = OFF_E8 + (size_t)kLTXT * kD;
    constexpr size_t OFF_QF  = OFF_W8 + 8 * SZ_W;
    constexpr size_t OFF_KF  = OFF_QF + (size_t)kS * kD * 4;
    constexpr size_t OFF_VF  = OFF_KF + (size_t)kS * kD * 4;
    constexpr size_t OFF_QH  = OFF_VF + (size_t)kS * kD * 4;
    constexpr size_t OFF_KH  = OFF_QH + (size_t)kS * kD * 2;
    constexpr size_t OFF_VT  = OFF_KH + (size_t)kS * kD * 2;
    constexpr size_t OFF_AO  = OFF_QF;
    constexpr size_t OFF_O8  = OFF_VF;

    char* ws = (char*)d_ws;
    unsigned*    amax   = (unsigned*)ws;
    float*       wscale = (float*)(ws + OFF_WS);
    signed char* H8     = (signed char*)(ws + OFF_H8);
    signed char* E8     = (signed char*)(ws + OFF_E8);
    signed char* W8     = (signed char*)(ws + OFF_W8);
    float*       QF     = (float*)(ws + OFF_QF);
    float*       KF     = (float*)(ws + OFF_KF);
    float*       VF     = (float*)(ws + OFF_VF);
    _Float16*    QHh    = (_Float16*)(ws + OFF_QH);
    _Float16*    KHh    = (_Float16*)(ws + OFF_KH);
    _Float16*    VT     = (_Float16*)(ws + OFF_VT);
    float*       AO     = (float*)(ws + OFF_AO);
    signed char* O8     = (signed char*)(ws + OFF_O8);

    hipMemsetAsync(amax, 0, 256, stream);

    auto nblk = [](long n) { return (unsigned)((n + 255) / 256); };

    absmax_kernel<<<nblk((long)kLIMG * kD), 256, 0, stream>>>(
        hidden, kLIMG * kD, 0, amax + 0, amax + 0);
    absmax_kernel<<<nblk((long)kLTXT * kD), 256, 0, stream>>>(
        encoder, kLTXT * kD, 0, amax + 1, amax + 1);
    quant_i8_kernel<<<nblk((long)kLIMG * kD), 256, 0, stream>>>(
        hidden, amax + 0, H8, kLIMG * kD);
    quant_i8_kernel<<<nblk((long)kLTXT * kD), 256, 0, stream>>>(
        encoder, amax + 1, E8, kLTXT * kD);

    for (int m = 0; m < 8; ++m)
        quant_w_kernel<<<kD / 8, 256, 0, stream>>>(
            Wf[m], W8 + m * SZ_W, wscale + m * kD, kD, kD);

    const dim3 gEnc(kLTXT / 128, kD / 128), gImg(kLIMG / 128, kD / 128);
    gemm_i8_kernel<<<gEnc, 256, 0, stream>>>(E8, W8 + 3 * SZ_W, wscale + 3 * kD,
        Bf[3], amax + 1, QF, kLTXT, kD, kD);
    gemm_i8_kernel<<<gImg, 256, 0, stream>>>(H8, W8 + 0 * SZ_W, wscale + 0 * kD,
        Bf[0], amax + 0, QF + (size_t)kLTXT * kD, kLIMG, kD, kD);
    gemm_i8_kernel<<<gEnc, 256, 0, stream>>>(E8, W8 + 4 * SZ_W, wscale + 4 * kD,
        Bf[4], amax + 1, KF, kLTXT, kD, kD);
    gemm_i8_kernel<<<gImg, 256, 0, stream>>>(H8, W8 + 1 * SZ_W, wscale + 1 * kD,
        Bf[1], amax + 0, KF + (size_t)kLTXT * kD, kLIMG, kD, kD);
    gemm_i8_kernel<<<gEnc, 256, 0, stream>>>(E8, W8 + 5 * SZ_W, wscale + 5 * kD,
        Bf[5], amax + 1, VF, kLTXT, kD, kD);
    gemm_i8_kernel<<<gImg, 256, 0, stream>>>(H8, W8 + 2 * SZ_W, wscale + 2 * kD,
        Bf[2], amax + 0, VF + (size_t)kLTXT * kD, kLIMG, kD, kD);

    rmsnorm_absmax_kernel<<<(kS * kH) / 8, 256, 0, stream>>>(
        QF, norm_q, norm_aq, amax + 2, amax + 3, kS * kH);
    rmsnorm_absmax_kernel<<<(kS * kH) / 8, 256, 0, stream>>>(
        KF, norm_k, norm_ak, amax + 4, amax + 5, kS * kH);
    absmax_kernel<<<nblk((long)kS * kD), 256, 0, stream>>>(
        VF, kS * kD, kLTXT * kD, amax + 6, amax + 7);

    quant_rope_kernel<<<nblk((long)kS * (kD / 2)), 256, 0, stream>>>(
        QF, rope_cos, rope_sin, amax + 2, amax + 3, QHh);
    quant_rope_kernel<<<nblk((long)kS * (kD / 2)), 256, 0, stream>>>(
        KF, rope_cos, rope_sin, amax + 4, amax + 5, KHh);
    quant_v_kernel<<<nblk((long)kS * kD), 256, 0, stream>>>(
        VF, amax + 6, amax + 7, VT);

    flash_attn_kernel<<<dim3(kS / 64, kH), 128, 0, stream>>>(QHh, KHh, VT, AO);

    absmax_kernel<<<nblk((long)kS * kD), 256, 0, stream>>>(
        AO, kS * kD, 0, amax + 8, amax + 8);
    quant_i8_kernel<<<nblk((long)kS * kD), 256, 0, stream>>>(
        AO, amax + 8, O8, kS * kD);

    float* out_img = (float*)d_out;
    float* out_enc = (float*)d_out + (size_t)kLIMG * kD;
    gemm_i8_kernel<<<gImg, 256, 0, stream>>>(O8 + (size_t)kLTXT * kD,
        W8 + 6 * SZ_W, wscale + 6 * kD, Bf[6], amax + 8, out_img, kLIMG, kD, kD);
    gemm_i8_kernel<<<gEnc, 256, 0, stream>>>(O8,
        W8 + 7 * SZ_W, wscale + 7 * kD, Bf[7], amax + 8, out_enc, kLTXT, kD, kD);
}
